// RotarySelfAttention_76940044141108
// MI455X (gfx1250) — matrix operand order repaired and checked by reading, not compiled, not run on hardware
//
#include <hip/hip_runtime.h>
#include <hip/hip_bf16.h>
#include <math.h>
#include <stdint.h>

#define D_MODEL 2048
#define N_HEADS 16
#define N_KV 4
#define HEAD_DIM 128
#define SEQ_B 2
#define SEQ_L 2048
#define ROWS (SEQ_B*SEQ_L)                  // 4096
#define KV_COLS (N_KV*HEAD_DIM)             // 512
#define N_QKV (D_MODEL + 2*KV_COLS)         // 3072

typedef __attribute__((ext_vector_type(16))) __bf16 bf16x16;
typedef __attribute__((ext_vector_type(8)))  __bf16 bf16x8;
typedef __attribute__((ext_vector_type(8)))  float  f32x8;
typedef __attribute__((ext_vector_type(4)))  unsigned int u32x4;
typedef __attribute__((ext_vector_type(4)))  int i32x4;
typedef __attribute__((ext_vector_type(8)))  int i32x8;

#define WMMA_BF16(a,b,c) __builtin_amdgcn_wmma_f32_16x16x32_bf16(false,(a),false,(b),(short)0,(c),false,false)

// A-fragment (16x32 bf16, MxK): lane = row (lane&15), 16 contiguous K at (lane>>4)*16.
static __device__ __forceinline__ bf16x16 frag_a(const __bf16* base, int ldk, int lane) {
  int row = lane & 15;
  int ko  = (lane >> 4) << 4;
  const __bf16* p = base + (size_t)row * ldk + ko;
  union { bf16x16 v; bf16x8 h[2]; } u;
  u.h[0] = *(const bf16x8*)(p);
  u.h[1] = *(const bf16x8*)(p + 8);
  return u.v;
}
// B-fragment (32x16 bf16, KxN): lane = column (lane&15), 16 contiguous K at (lane>>4)*16.
// Source must be N-major (row = N, contiguous along K), i.e. W transposed.
static __device__ __forceinline__ bf16x16 frag_b(const __bf16* base, int ldk, int lane) {
  int col = lane & 15;
  int kb  = (lane >> 4) << 4;
  const __bf16* p = base + (size_t)col * ldk + kb;
  union { bf16x16 v; bf16x8 h[2]; } u;
  u.h[0] = *(const bf16x8*)(p);
  u.h[1] = *(const bf16x8*)(p + 8);
  return u.v;
}

// ---- Tensor Data Mover: DMA a 2D bf16 tile (tile_rows x tile_k elems, row
// stride ld_elems) from global memory into LDS at byte address lds_addr.
// D# layout per CDNA5 ISA; groups 2/3 zero (2D).
// This toolchain declares the 6-arg builtin: (g0, g1, g2, g3, g4, cpol).
static __device__ __forceinline__ void tdm_load_2d(unsigned lds_addr, const void* gptr,
                                                   unsigned tile_k, unsigned tile_rows,
                                                   unsigned ld_elems) {
  unsigned long long ga = (unsigned long long)(uintptr_t)gptr;
  unsigned td0 = ld_elems;            // tensor dim0 extent (elements) - no OOB in our use
  unsigned td1 = 1u << 20;            // generous dim1 extent
  unsigned long long s0 = ld_elems;   // dim0 stride (elements)
  u32x4 g0;
  g0[0] = 1u;                                                // count=1, user descriptor
  g0[1] = lds_addr;                                          // bits 63:32  lds_addr
  g0[2] = (unsigned)(ga & 0xFFFFFFFFu);                      // bits 95:64  global_addr lo
  g0[3] = (unsigned)((ga >> 32) & 0x01FFFFFFu) | (2u << 30); // addr[56:32] | type=2
  i32x8 g1;
  g1[0] = (int)(1u << 16);                                   // wg_mask=0, data_size=1 (2B)
  g1[1] = (int)((td0 & 0xFFFFu) << 16);                      // tensor_dim0[15:0] @ bit48
  g1[2] = (int)(((td0 >> 16) & 0xFFFFu) | ((td1 & 0xFFFFu) << 16));
  g1[3] = (int)(((td1 >> 16) & 0xFFFFu) | ((tile_k & 0xFFFFu) << 16)); // tile_dim0
  g1[4] = (int)(tile_rows & 0xFFFFu);                        // tile_dim1 (tile_dim2=0)
  g1[5] = (int)(s0 & 0xFFFFFFFFu);                           // tensor_dim0_stride lo
  g1[6] = (int)((s0 >> 32) & 0xFFFFu);                       // stride hi, dim1_stride=0
  g1[7] = 0;
  i32x4 z4 = {};
  i32x8 z8 = {};
  __builtin_amdgcn_tensor_load_to_lds(g0, g1, z4, z4, z8, 0);
}

__global__ void cvt_f32_bf16(const float* __restrict__ in, __bf16* __restrict__ out, int n) {
  int i = blockIdx.x * blockDim.x + threadIdx.x;
  if (i < n) out[i] = (__bf16)in[i];
}

// in: K x N row-major (f32); out: N x K row-major (bf16)
__global__ void transpose_f32_bf16(const float* __restrict__ in, __bf16* __restrict__ out, int K, int N) {
  int i = blockIdx.x * blockDim.x + threadIdx.x;
  if (i >= K * N) return;
  int k = i % K, n = i / K;
  out[i] = (__bf16)in[(size_t)k * N + n];
}

// C[M,N](f32) = A[M,K](bf16 row-major) * BT[N,K](bf16 N-major)^T.
// Workgroup (4 waves) computes a 64x64 tile; A/B 64x64 slabs staged to LDS by
// the Tensor Data Mover each K-step; each wave computes a 32x32 quadrant.
__global__ void __launch_bounds__(128) gemm_bf16_tdm(const __bf16* __restrict__ A,
                                                     const __bf16* __restrict__ BT,
                                                     float* __restrict__ C, int M, int N, int K) {
  __shared__ __bf16 ldsA[64 * 64];
  __shared__ __bf16 ldsB[64 * 64];
  int lane = threadIdx.x & 31;
  int wave = threadIdx.x >> 5;
  int ntn  = N >> 6;
  int tn = blockIdx.x % ntn, tm = blockIdx.x / ntn;
  int m0 = tm << 6, n0 = tn << 6;
  int mq = (wave >> 1) << 5;          // 0 or 32: M quadrant within tile
  int nq = (wave & 1) << 5;           // 0 or 32: N quadrant within tile
  unsigned la = (unsigned)(uintptr_t)&ldsA[0];
  unsigned lb = (unsigned)(uintptr_t)&ldsB[0];

  f32x8 acc[2][2] = {};
  for (int k0 = 0; k0 < K; k0 += 64) {
    __syncthreads();                  // previous compute done before overwrite
    if (wave == 0) {
      tdm_load_2d(la, A  + (size_t)m0 * K + k0, 64, 64, (unsigned)K);
      tdm_load_2d(lb, BT + (size_t)n0 * K + k0, 64, 64, (unsigned)K);
      __builtin_amdgcn_s_wait_tensorcnt(0);
    }
    __syncthreads();                  // slabs visible to all waves
#pragma unroll
    for (int ks = 0; ks < 64; ks += 32) {
      bf16x16 a0 = frag_a(ldsA + (size_t)(mq +  0) * 64 + ks, 64, lane);
      bf16x16 a1 = frag_a(ldsA + (size_t)(mq + 16) * 64 + ks, 64, lane);
      bf16x16 b0 = frag_b(ldsB + (size_t)(nq +  0) * 64 + ks, 64, lane);
      bf16x16 b1 = frag_b(ldsB + (size_t)(nq + 16) * 64 + ks, 64, lane);
      acc[0][0] = WMMA_BF16(a0, b0, acc[0][0]);
      acc[0][1] = WMMA_BF16(a0, b1, acc[0][1]);
      acc[1][0] = WMMA_BF16(a1, b0, acc[1][0]);
      acc[1][1] = WMMA_BF16(a1, b1, acc[1][1]);
    }
  }
  int col = lane & 15, hl = lane >> 4;
#pragma unroll
  for (int i = 0; i < 2; ++i)
#pragma unroll
    for (int j = 0; j < 2; ++j)
#pragma unroll
      for (int r = 0; r < 8; ++r)
        C[(size_t)(m0 + mq + 16 * i + 8 * hl + r) * N + (n0 + nq + 16 * j + col)] = acc[i][j][r];
}

// RoPE + QK-RMS for Q; one wave per (b,l,head). proj rows are [b*L+l][3072].
__global__ void rope_rms_q(const float* __restrict__ proj, const float* __restrict__ cosb,
                           const float* __restrict__ sinb, __bf16* __restrict__ qb) {
  int lane = threadIdx.x & 31;
  int wid  = blockIdx.x * (blockDim.x >> 5) + (threadIdx.x >> 5);
  int head = wid % N_HEADS;
  int row  = wid / N_HEADS;                 // b*L + l
  const float* t  = proj + (size_t)row * N_QKV + head * HEAD_DIM;
  const float* cp = cosb + ((size_t)row * N_HEADS + head) * HEAD_DIM;
  const float* sp = sinb + ((size_t)row * N_HEADS + head) * HEAD_DIM;
  float t0 = t[lane], t1 = t[lane + 32], t2 = t[lane + 64], t3 = t[lane + 96];
  float c0 = cp[lane], s0 = sp[lane], c1 = cp[lane + 32], s1 = sp[lane + 32];
  float r0 = t0 * c0 - t2 * s0;
  float r1 = t1 * c1 - t3 * s1;
  float r2 = t2 * c0 + t0 * s0;
  float r3 = t3 * c1 + t1 * s1;
  float ms = r0 * r0 + r1 * r1 + r2 * r2 + r3 * r3;
#pragma unroll
  for (int m = 16; m >= 1; m >>= 1) ms += __shfl_xor(ms, m, 32);
  float sc = rsqrtf(ms * (1.0f / HEAD_DIM) + 1e-6f);
  int b = row / SEQ_L, l = row % SEQ_L;
  __bf16* o = qb + (((size_t)b * N_HEADS + head) * SEQ_L + l) * HEAD_DIM;
  o[lane]      = (__bf16)(r0 * sc);
  o[lane + 32] = (__bf16)(r1 * sc);
  o[lane + 64] = (__bf16)(r2 * sc);
  o[lane + 96] = (__bf16)(r3 * sc);
}

__global__ void rope_rms_k(const float* __restrict__ proj, const float* __restrict__ cosb,
                           const float* __restrict__ sinb, __bf16* __restrict__ kb) {
  int lane = threadIdx.x & 31;
  int wid  = blockIdx.x * (blockDim.x >> 5) + (threadIdx.x >> 5);
  int head = wid % N_KV;                    // k uses cos[:, :, :N_KV, :]
  int row  = wid / N_KV;
  const float* t  = proj + (size_t)row * N_QKV + D_MODEL + head * HEAD_DIM;
  const float* cp = cosb + ((size_t)row * N_HEADS + head) * HEAD_DIM;
  const float* sp = sinb + ((size_t)row * N_HEADS + head) * HEAD_DIM;
  float t0 = t[lane], t1 = t[lane + 32], t2 = t[lane + 64], t3 = t[lane + 96];
  float c0 = cp[lane], s0 = sp[lane], c1 = cp[lane + 32], s1 = sp[lane + 32];
  float r0 = t0 * c0 - t2 * s0;
  float r1 = t1 * c1 - t3 * s1;
  float r2 = t2 * c0 + t0 * s0;
  float r3 = t3 * c1 + t1 * s1;
  float ms = r0 * r0 + r1 * r1 + r2 * r2 + r3 * r3;
#pragma unroll
  for (int m = 16; m >= 1; m >>= 1) ms += __shfl_xor(ms, m, 32);
  float sc = rsqrtf(ms * (1.0f / HEAD_DIM) + 1e-6f);
  int b = row / SEQ_L, l = row % SEQ_L;
  __bf16* o = kb + (((size_t)b * N_KV + head) * SEQ_L + l) * HEAD_DIM;
  o[lane]      = (__bf16)(r0 * sc);
  o[lane + 32] = (__bf16)(r1 * sc);
  o[lane + 64] = (__bf16)(r2 * sc);
  o[lane + 96] = (__bf16)(r3 * sc);
}

// v[b][l][kv][d] (inside proj) -> vT[b][kv][d][l] bf16
__global__ void vtrans(const float* __restrict__ proj, __bf16* __restrict__ vT) {
  int i = blockIdx.x * blockDim.x + threadIdx.x;
  const int total = SEQ_B * N_KV * HEAD_DIM * SEQ_L;
  if (i >= total) return;
  int l  = i % SEQ_L;
  int d  = (i / SEQ_L) % HEAD_DIM;
  int kv = (i / (SEQ_L * HEAD_DIM)) % N_KV;
  int b  = i / (SEQ_L * HEAD_DIM * N_KV);
  vT[i] = (__bf16)proj[(size_t)(b * SEQ_L + l) * N_QKV + D_MODEL + KV_COLS + kv * HEAD_DIM + d];
}

// Flash attention: one wave per (b, head, 16-query tile). Computes S^T = K*Q^T,
// online softmax over key blocks of 32, O^T += V^T * P^T. Output bf16 [b][l][h*128+d].
__global__ void __launch_bounds__(128) flash_attn(const __bf16* __restrict__ qb,
                                                  const __bf16* __restrict__ kb,
                                                  const __bf16* __restrict__ vT,
                                                  const int* __restrict__ mask,
                                                  __bf16* __restrict__ ob) {
  const float scale = 0.08838834764831845f;  // 1/sqrt(128)
  int lane = threadIdx.x & 31;
  int wid  = blockIdx.x * (blockDim.x >> 5) + (threadIdx.x >> 5);
  int qt   = wid & 127;
  int head = (wid >> 7) & 15;
  int b    = wid >> 11;
  int q0   = qt << 4;
  int kv   = head >> 2;                      // jnp.repeat(k, 4, axis=2)
  int col = lane & 15, hl = lane >> 4;
  int qpos = q0 + col;

  const __bf16* qbase = qb + (((size_t)b * N_HEADS + head) * SEQ_L + q0) * HEAD_DIM;
  const __bf16* kbase = kb + (((size_t)b * N_KV + kv) * SEQ_L) * HEAD_DIM;
  const __bf16* vbase = vT + (((size_t)b * N_KV + kv) * HEAD_DIM) * SEQ_L;
  const int* mrow = mask + b * SEQ_L;

  bf16x16 qf[4];
#pragma unroll
  for (int kc = 0; kc < 4; ++kc) qf[kc] = frag_b(qbase + kc * 32, HEAD_DIM, lane);

  f32x8 acc[8] = {};
  float m_i = -INFINITY, l_i = 0.0f;

  for (int m0 = 0; m0 < q0 + 16; m0 += 32) {
    f32x8 s[2] = {};
#pragma unroll
    for (int t = 0; t < 2; ++t)
#pragma unroll
      for (int kc = 0; kc < 4; ++kc) {
        bf16x16 a = frag_a(kbase + (size_t)(m0 + 16 * t) * HEAD_DIM + kc * 32, HEAD_DIM, lane);
        s[t] = WMMA_BF16(a, qf[kc], s[t]);
      }
    __builtin_prefetch(kbase + (size_t)(m0 + 32) * HEAD_DIM, 0, 3);

    // mask + scale, per-query (column) block max across both lane halves
    float sv[2][8];
    float bm = -INFINITY;
#pragma unroll
    for (int t = 0; t < 2; ++t)
#pragma unroll
      for (int r = 0; r < 8; ++r) {
        int mpos = m0 + 16 * t + 8 * hl + r;
        float v = s[t][r] * scale;
        if (!mrow[mpos]) v -= 10000.0f;     // key mask
        if (mpos > qpos) v -= 10000.0f;     // causal
        sv[t][r] = v;
        bm = fmaxf(bm, v);
      }
    bm = fmaxf(bm, __shfl_xor(bm, 16, 32));
    float m_new = fmaxf(m_i, bm);
    float p[2][8], ls = 0.0f;
#pragma unroll
    for (int t = 0; t < 2; ++t)
#pragma unroll
      for (int r = 0; r < 8; ++r) { p[t][r] = __expf(sv[t][r] - m_new); ls += p[t][r]; }
    ls += __shfl_xor(ls, 16, 32);
    float sf = __expf(m_i - m_new);
    l_i = l_i * sf + ls;
    m_i = m_new;
#pragma unroll
    for (int dt = 0; dt < 8; ++dt)
#pragma unroll
      for (int r = 0; r < 8; ++r) acc[dt][r] = acc[dt][r] * sf;

    // Rebuild P^T (32x16) as a B-fragment: lane needs K rows (lane>>4)*16..+15 of its column.
    bf16x16 pf;
#pragma unroll
    for (int r = 0; r < 8; ++r) {
      float o0 = __shfl_xor(p[0][r], 16, 32);
      float o1 = __shfl_xor(p[1][r], 16, 32);
      pf[r]     = (__bf16)(hl ? o1 : p[0][r]);
      pf[8 + r] = (__bf16)(hl ? p[1][r] : o0);
    }
#pragma unroll
    for (int dt = 0; dt < 8; ++dt) {
      bf16x16 a = frag_a(vbase + (size_t)(dt * 16) * SEQ_L + m0, SEQ_L, lane);
      acc[dt] = WMMA_BF16(a, pf, acc[dt]);
    }
  }

  float inv = 1.0f / l_i;
  __bf16* orow = ob + (size_t)(b * SEQ_L + qpos) * D_MODEL + head * HEAD_DIM;
#pragma unroll
  for (int dt = 0; dt < 8; ++dt) {
    bf16x8 w;
#pragma unroll
    for (int r = 0; r < 8; ++r) w[r] = (__bf16)(acc[dt][r] * inv);
    *(bf16x8*)(orow + dt * 16 + 8 * hl) = w;
  }
}

extern "C" void kernel_launch(void* const* d_in, const int* in_sizes, int n_in,
                              void* d_out, int out_size, void* d_ws, size_t ws_size,
                              hipStream_t stream) {
  const float* x    = (const float*)d_in[0];
  const float* cosb = (const float*)d_in[1];
  const float* sinb = (const float*)d_in[2];
  const int*   mask = (const int*)d_in[3];
  const float* Wq   = (const float*)d_in[4];
  const float* Wk   = (const float*)d_in[5];
  const float* Wv   = (const float*)d_in[6];
  const float* Wo   = (const float*)d_in[7];
  float* out = (float*)d_out;

  char* ws = (char*)d_ws;
  size_t off = 0;
  auto alloc = [&](size_t bytes) -> void* {
    void* p = ws + off;
    off += (bytes + 255) & ~(size_t)255;
    return p;
  };
  __bf16* xb    = (__bf16*)alloc((size_t)ROWS * D_MODEL * 2);
  __bf16* WqkvT = (__bf16*)alloc((size_t)N_QKV * D_MODEL * 2);
  __bf16* WoT   = (__bf16*)alloc((size_t)D_MODEL * D_MODEL * 2);
  float*  proj  = (float*)alloc((size_t)ROWS * N_QKV * 4);
  __bf16* qbuf  = (__bf16*)alloc((size_t)SEQ_B * N_HEADS * SEQ_L * HEAD_DIM * 2);
  __bf16* kbuf  = (__bf16*)alloc((size_t)SEQ_B * N_KV * SEQ_L * HEAD_DIM * 2);
  __bf16* vtb   = (__bf16*)alloc((size_t)SEQ_B * N_KV * HEAD_DIM * SEQ_L * 2);
  __bf16* obuf  = (__bf16*)alloc((size_t)ROWS * D_MODEL * 2);

  int n = ROWS * D_MODEL;
  cvt_f32_bf16<<<(n + 255) / 256, 256, 0, stream>>>(x, xb, n);
  n = D_MODEL * D_MODEL;
  transpose_f32_bf16<<<(n + 255) / 256, 256, 0, stream>>>(Wq, WqkvT, D_MODEL, D_MODEL);
  n = D_MODEL * KV_COLS;
  transpose_f32_bf16<<<(n + 255) / 256, 256, 0, stream>>>(
      Wk, WqkvT + (size_t)D_MODEL * D_MODEL, D_MODEL, KV_COLS);
  transpose_f32_bf16<<<(n + 255) / 256, 256, 0, stream>>>(
      Wv, WqkvT + (size_t)(D_MODEL + KV_COLS) * D_MODEL, D_MODEL, KV_COLS);
  n = D_MODEL * D_MODEL;
  transpose_f32_bf16<<<(n + 255) / 256, 256, 0, stream>>>(Wo, WoT, D_MODEL, D_MODEL);

  {
    int blocks = (ROWS / 64) * (N_QKV / 64);   // 64x48
    gemm_bf16_tdm<<<blocks, 128, 0, stream>>>(xb, WqkvT, proj, ROWS, N_QKV, D_MODEL);
  }
  {
    int waves = SEQ_B * SEQ_L * N_HEADS;
    rope_rms_q<<<waves / 8, 256, 0, stream>>>(proj, cosb, sinb, qbuf);
  }
  {
    int waves = SEQ_B * SEQ_L * N_KV;
    rope_rms_k<<<waves / 8, 256, 0, stream>>>(proj, cosb, sinb, kbuf);
  }
  {
    int tot = SEQ_B * N_KV * HEAD_DIM * SEQ_L;
    vtrans<<<(tot + 255) / 256, 256, 0, stream>>>(proj, vtb);
  }
  {
    int waves = SEQ_B * N_HEADS * (SEQ_L / 16);
    flash_attn<<<waves / 4, 128, 0, stream>>>(qbuf, kbuf, vtb, mask, obuf);
  }
  {
    int blocks = (ROWS / 64) * (D_MODEL / 64); // 64x32
    gemm_bf16_tdm<<<blocks, 128, 0, stream>>>(obuf, WoT, out, ROWS, D_MODEL, D_MODEL);
  }
}
